// ContrastiveLoss_53266184405266
// MI455X (gfx1250) — matrix operand order repaired and checked by reading, not compiled, not run on hardware
//
#include <hip/hip_runtime.h>

typedef _Float16 v16h __attribute__((ext_vector_type(16)));
typedef _Float16 h8   __attribute__((ext_vector_type(8)));
typedef float    v8f  __attribute__((ext_vector_type(8)));

#define N_TOT      8192
#define NS         4096
#define D          512
#define ROW_BLK    128                    // 8 waves x 16 rows
#define COL_SPLIT  4
#define COLS_PB    (N_TOT / COL_SPLIT)    // 2048 columns per block
#define NCT        (COLS_PB / 16)         // 128 column tiles per block
#define BSTRIDE    520                    // halfs per LDS B row (+16B pad)
#define INV_T      2.0f

#if defined(__has_builtin)
#  if __has_builtin(__builtin_amdgcn_s_wait_asynccnt)
#    define WAIT_ASYNC() __builtin_amdgcn_s_wait_asynccnt(0)
#  endif
#endif
#ifndef WAIT_ASYNC
#  define WAIT_ASYNC() asm volatile("s_wait_asynccnt 0" ::: "memory")
#endif

// 16B async copy global -> LDS (no VGPR data path, tracked by ASYNCcnt).
// VGLOBAL async form: VDST = LDS byte address VGPR, VADDR = 64-bit address
// pair, SADDR = off. Generic LDS pointers carry the LDS offset in their low
// 32 bits.
__device__ __forceinline__ void async_cp16(const _Float16* g, _Float16* l) {
  const unsigned lofs            = (unsigned)(uintptr_t)l;
  const unsigned long long gaddr = (unsigned long long)(uintptr_t)g;
  asm volatile("global_load_async_to_lds_b128 %0, %1, off"
               :: "v"(lofs), "v"(gaddr) : "memory");
}

// ---------------------------------------------------------------- normalize
__global__ __launch_bounds__(256)
void normalize_kernel(const float* __restrict__ x1, const float* __restrict__ x2,
                      _Float16* __restrict__ y) {
  __shared__ float wsum[8];
  const int row = blockIdx.x;
  const float* src = (row < NS) ? (x1 + (size_t)row * D)
                                : (x2 + (size_t)(row - NS) * D);
  const int tid = threadIdx.x;
  float v0 = src[tid];
  float v1 = src[tid + 256];
  float ss = v0 * v0 + v1 * v1;
  #pragma unroll
  for (int off = 16; off; off >>= 1) ss += __shfl_down(ss, off);
  if ((tid & 31) == 0) wsum[tid >> 5] = ss;
  __syncthreads();
  if (tid == 0) {
    float t = 0.f;
    #pragma unroll
    for (int i = 0; i < 8; ++i) t += wsum[i];
    wsum[0] = 1.0f / sqrtf(t);
  }
  __syncthreads();
  const float r = wsum[0];
  _Float16* dst = y + (size_t)row * D;
  dst[tid]       = (_Float16)(v0 * r);
  dst[tid + 256] = (_Float16)(v1 * r);
}

// --------------------------------------------- fused GEMM + row-partial sums
__global__ __launch_bounds__(256, 1)
void simgemm_kernel(const _Float16* __restrict__ y,
                    float* __restrict__ rowE, float* __restrict__ rowS,
                    float* __restrict__ rowP) {
  __shared__ _Float16 ldsB[2][16][BSTRIDE];   // double-buffered B tile, ~33 KB

  const int tid  = threadIdx.x;
  const int lane = tid & 31;
  const int wv   = tid >> 5;
  const int l15  = lane & 15;
  const int half = lane >> 4;

  const int rb       = blockIdx.x >> 2;       // 64 row blocks
  const int cs       = blockIdx.x & 3;        // 4 column splits
  const int row_base = rb * ROW_BLK;
  const int col_base = cs * COLS_PB;

  // --- preload this wave's A fragments (16 rows x full K) into registers ---
  const int arow = row_base + wv * 16 + l15;
  const _Float16* abase = y + (size_t)arow * D + half * 8;
  v16h Afrag[16];
  #pragma unroll
  for (int kk = 0; kk < 16; ++kk) {
    h8 lo = *(const h8*)(abase + kk * 32);
    h8 hi = *(const h8*)(abase + kk * 32 + 16);
    #pragma unroll
    for (int t = 0; t < 8; ++t) { Afrag[kk][t] = lo[t]; Afrag[kk][8 + t] = hi[t]; }
  }

  float accE[8], accS[8];
  #pragma unroll
  for (int v = 0; v < 8; ++v) { accE[v] = 0.f; accS[v] = 0.f; }

  // staging geometry: thread -> (row-in-tile, 8-half chunk)
  const int c8 = tid & 63;
  const int r0 = tid >> 6;

  // prologue: async-stage tile 0 into buffer 0
  #pragma unroll
  for (int q = 0; q < 4; ++q)
    async_cp16(y + (size_t)(col_base + r0 + 4 * q) * D + c8 * 8,
               &ldsB[0][r0 + 4 * q][c8 * 8]);
  WAIT_ASYNC();
  __syncthreads();

  for (int ctl = 0; ctl < NCT; ++ctl) {
    const int buf = ctl & 1;
    const int n0  = col_base + ctl * 16;

    // kick off async copy of the next tile into the other buffer;
    // it lands in LDS while we run the 16-WMMA burst below
    if (ctl + 1 < NCT) {
      #pragma unroll
      for (int q = 0; q < 4; ++q)
        async_cp16(y + (size_t)(n0 + 16 + r0 + 4 * q) * D + c8 * 8,
                   &ldsB[buf ^ 1][r0 + 4 * q][c8 * 8]);
    }

    // 16 WMMAs over K=512; A in registers, B from LDS, one stage ahead
    const _Float16* bb = &ldsB[buf][l15][half * 8];
    v16h bcur, bnxt;
    {
      h8 b0 = *(const h8*)(bb);
      h8 b1 = *(const h8*)(bb + 16);
      #pragma unroll
      for (int t = 0; t < 8; ++t) { bcur[t] = b0[t]; bcur[8 + t] = b1[t]; }
    }
    v8f c = {};
    #pragma unroll
    for (int kk = 0; kk < 16; ++kk) {
      if (kk < 15) {
        h8 b0 = *(const h8*)(bb + (kk + 1) * 32);
        h8 b1 = *(const h8*)(bb + (kk + 1) * 32 + 16);
        #pragma unroll
        for (int t = 0; t < 8; ++t) { bnxt[t] = b0[t]; bnxt[8 + t] = b1[t]; }
      }
      c = __builtin_amdgcn_wmma_f32_16x16x32_f16(false, Afrag[kk], false, bcur,
                                                 (short)0, c, false, false);
      bcur = bnxt;
    }

    // fused epilogue: element v -> row = wv*16 + v + 8*half, col = n0 + l15
    const int j = n0 + l15;
    #pragma unroll
    for (int v = 0; v < 8; ++v) {
      const int i       = row_base + wv * 16 + v + half * 8;
      const float logit = c[v] * INV_T;
      if (j == i) {
        // diagonal: masked from pos and neg (its exp(0) added in phase 2)
      } else if (j == (i ^ NS)) {
        rowP[i] = logit;                  // globally unique writer per row
      } else {
        accE[v] += __expf(logit);
        accS[v] += logit;
      }
    }

    WAIT_ASYNC();                         // my async writes into buf^1 landed
    __syncthreads();                      // everyone's landed; reads done too
  }

  // per-row reduce across the 16 lanes of each half (rows disjoint per group)
  #pragma unroll
  for (int v = 0; v < 8; ++v) {
    #pragma unroll
    for (int off = 8; off; off >>= 1) {
      accE[v] += __shfl_xor(accE[v], off);
      accS[v] += __shfl_xor(accS[v], off);
    }
  }
  if (l15 == 0) {
    #pragma unroll
    for (int v = 0; v < 8; ++v) {
      const int i = row_base + wv * 16 + v + half * 8;
      atomicAdd(&rowE[i], accE[v]);       // one atomic per (row, col-split)
      atomicAdd(&rowS[i], accS[v]);
    }
  }
}

// ------------------------------------------- per-row log-sum-exp + reduction
__global__ __launch_bounds__(256)
void loss_kernel(const float* __restrict__ rowE, const float* __restrict__ rowS,
                 const float* __restrict__ rowP, float* __restrict__ gacc) {
  __shared__ float red[3][8];
  const int i = blockIdx.x * 256 + threadIdx.x;
  const float E = rowE[i] + 2.0f;         // exp(0) at the two masked slots
  const float p = rowP[i];
  float loss = -(p - logf(__expf(p) + E));
  float ps   = p;
  float ns   = rowS[i];
  #pragma unroll
  for (int off = 16; off; off >>= 1) {
    loss += __shfl_down(loss, off);
    ps   += __shfl_down(ps, off);
    ns   += __shfl_down(ns, off);
  }
  const int lane = threadIdx.x & 31, w = threadIdx.x >> 5;
  if (lane == 0) { red[0][w] = loss; red[1][w] = ps; red[2][w] = ns; }
  __syncthreads();
  if (threadIdx.x == 0) {
    float a = 0.f, b = 0.f, c = 0.f;
    #pragma unroll
    for (int k = 0; k < 8; ++k) { a += red[0][k]; b += red[1][k]; c += red[2][k]; }
    atomicAdd(&gacc[0], a);
    atomicAdd(&gacc[1], b);
    atomicAdd(&gacc[2], c);
  }
}

// ----------------------------------------------------------------- finalize
__global__ void finalize_kernel(const float* __restrict__ gacc, float* __restrict__ out) {
  out[0] = gacc[0] / 8192.0f;             // pos_sum = N
  out[1] = gacc[1] / 8192.0f;
  out[2] = gacc[2] / (8192.0f * 8190.0f); // neg_sum = N*(N-2)
}

extern "C" void kernel_launch(void* const* d_in, const int* in_sizes, int n_in,
                              void* d_out, int out_size, void* d_ws, size_t ws_size,
                              hipStream_t stream) {
  const float* x1 = (const float*)d_in[0];
  const float* x2 = (const float*)d_in[1];
  char* ws = (char*)d_ws;

  _Float16* y  = (_Float16*)ws;                                  // 8 MB
  size_t off   = (size_t)N_TOT * D * sizeof(_Float16);
  float* rowE  = (float*)(ws + off);                             // 32 KB
  float* rowS  = (float*)(ws + off + N_TOT * sizeof(float));     // 32 KB
  float* gacc  = (float*)(ws + off + 2 * N_TOT * sizeof(float)); // 16 B
  float* rowP  = (float*)(ws + off + 2 * N_TOT * sizeof(float) + 4 * sizeof(float));
  float* out   = (float*)d_out;

  (void)hipMemsetAsync(rowE, 0, 2 * N_TOT * sizeof(float) + 4 * sizeof(float), stream);
  normalize_kernel<<<N_TOT, 256, 0, stream>>>(x1, x2, y);
  simgemm_kernel<<<(N_TOT / ROW_BLK) * COL_SPLIT, 256, 0, stream>>>(y, rowE, rowS, rowP);
  loss_kernel<<<N_TOT / 256, 256, 0, stream>>>(rowE, rowS, rowP, gacc);
  finalize_kernel<<<1, 1, 0, stream>>>(gacc, out);
}
